// RawAttention_79783312491266
// MI455X (gfx1250) — matrix operand order repaired and checked by reading, not compiled, not run on hardware
//
#include <hip/hip_runtime.h>
#include <hip/hip_bf16.h>

typedef __attribute__((ext_vector_type(16))) _Float16 v16h;
typedef __attribute__((ext_vector_type(8)))  float    v8f;

union Frag { unsigned int u[8]; v16h h; };

// A-matrix 16x32 f16 fragment: lane m = lane&15, g = lane>>4.
// VGPR v holds K = (v>>2)*16 + g*8 + (v&3)*2 (+1)  -> dword index = (v>>2)*8 + g*4 + (v&3)
__device__ __forceinline__ v16h load_a_frag(const _Float16* row, int d, int g) {
    const unsigned int* p = (const unsigned int*)(row + d);
    Frag f;
#pragma unroll
    for (int v = 0; v < 8; ++v)
        f.u[v] = p[((v >> 2) << 3) + (g << 2) + (v & 3)];
    return f.h;
}

// B-matrix 32x16 f16 fragment: lane n = lane&15, g = lane>>4.
// VGPR v holds K = (v>>2)*16 + g*8 + (v&3)*2 (+1)  -> dword index = (v>>2)*8 + g*4 + (v&3)
__device__ __forceinline__ v16h load_b_frag(const _Float16* row, int d, int g) {
    const unsigned int* p = (const unsigned int*)(row + d);
    Frag f;
#pragma unroll
    for (int v = 0; v < 8; ++v)
        f.u[v] = p[((v >> 2) << 3) + (g << 2) + (v & 3)];
    return f.h;
}

__device__ __forceinline__ v8f wmma_f16(v16h a, v16h b, v8f c) {
    return __builtin_amdgcn_wmma_f32_16x16x32_f16(false, a, false, b,
                                                  (short)0, c, false, false);
}

__global__ void cvt_f32_f16(const float* __restrict__ s, _Float16* __restrict__ d, int n) {
    int i = blockIdx.x * blockDim.x + threadIdx.x;
    if (i < n) d[i] = (_Float16)s[i];
}

// 4 waves: w0 = q cols [c,c+16), w1 = q cols [c+256,c+272), w2/w3 same for k.
// Each wave: 16x16 GEMM tile over K=512, then RoPE using partner tile via LDS.
__global__ __launch_bounds__(128) void qk_proj_rope(
    const _Float16* __restrict__ h_h, const _Float16* __restrict__ wq_h,
    const _Float16* __restrict__ wk_h, const float* __restrict__ cosb,
    const float* __restrict__ sinb, _Float16* __restrict__ qh,
    _Float16* __restrict__ kh)
{
    const int w    = threadIdx.x >> 5;     // 0..3
    const int lane = threadIdx.x & 31;
    const int g    = lane >> 4;
    const int n    = lane & 15;            // doubles as A-row (m) and B/C col (n)
    const int mat  = w >> 1;               // 0=q, 1=k
    const int half = w & 1;                // 0 = cols<256, 1 = cols>=256
    const int R    = blockIdx.y << 4;      // global row base (0..16383)
    const int c0   = (blockIdx.x << 4) + (half << 8);

    const _Float16* W    = mat ? wk_h : wq_h;
    _Float16*       outp = mat ? kh : qh;

    const _Float16* arow = h_h + (size_t)(R + n) * 512;
    const _Float16* brow = W   + (size_t)(c0 + n) * 512;

    v8f acc = {0.f, 0.f, 0.f, 0.f, 0.f, 0.f, 0.f, 0.f};
    for (int d = 0; d < 512; d += 32) {
        v16h a = load_a_frag(arow, d, g);
        v16h b = load_b_frag(brow, d, g);
        acc = wmma_f16(a, b, acc);
    }

    __shared__ float tile[4][16][17];
#pragma unroll
    for (int r = 0; r < 8; ++r) tile[w][r + (g << 3)][n] = acc[r];
    __syncthreads();

    // rotate_half: n<256 -> -x[n+256], n>=256 -> +x[n-256]
    const float sgn = half ? 1.0f : -1.0f;
#pragma unroll
    for (int r = 0; r < 8; ++r) {
        int   row  = R + r + (g << 3);
        int   s    = row & 4095;
        int   col  = c0 + n;
        float part = tile[w ^ 1][r + (g << 3)][n];
        float cs   = cosb[s * 512 + col];
        float sn   = sinb[s * 512 + col];
        float val  = acc[r] * cs + sgn * part * sn;
        outp[(size_t)row * 512 + col] = (_Float16)val;
    }
}

// One wave per 32x32 score block (4 WMMA tiles, A/B fragment reuse, 4 independent
// accumulator chains). Block = 8 waves = 64 rows x 128 cols.
__global__ __launch_bounds__(256) void scores_kernel(
    const _Float16* __restrict__ qh, const _Float16* __restrict__ kh,
    float* __restrict__ out)
{
    const int b    = blockIdx.z;
    const int w    = threadIdx.x >> 5;
    const int lane = threadIdx.x & 31;
    const int g    = lane >> 4;
    const int n    = lane & 15;
    const int Rw   = (blockIdx.y << 6) + ((w >> 2) << 5);  // query rows [Rw, Rw+32)
    const int Cw   = (blockIdx.x << 7) + ((w & 3) << 5);   // key cols  [Cw, Cw+32)
    if (Cw > Rw + 31) return;                              // fully above causal diagonal

    const _Float16* ar0 = qh + (size_t)(b * 4096 + Rw + n) * 512;
    const _Float16* ar1 = ar0 + (size_t)16 * 512;
    const _Float16* br0 = kh + (size_t)(b * 4096 + Cw + n) * 512;
    const _Float16* br1 = br0 + (size_t)16 * 512;

    v8f acc00 = {0.f, 0.f, 0.f, 0.f, 0.f, 0.f, 0.f, 0.f};
    v8f acc01 = acc00, acc10 = acc00, acc11 = acc00;

    for (int d = 0; d < 512; d += 32) {
        v16h A0 = load_a_frag(ar0, d, g);
        v16h A1 = load_a_frag(ar1, d, g);
        v16h B0 = load_b_frag(br0, d, g);
        v16h B1 = load_b_frag(br1, d, g);
        acc00 = wmma_f16(A0, B0, acc00);
        acc01 = wmma_f16(A0, B1, acc01);
        acc10 = wmma_f16(A1, B0, acc10);
        acc11 = wmma_f16(A1, B1, acc11);
    }

    const float scale = 0.04419417382415922f;              // 1/sqrt(512)
    float* o = out + ((size_t)b * 4096 + Rw) * 4096 + Cw;
#pragma unroll
    for (int r = 0; r < 8; ++r) {
        int row0 = r + (g << 3);
        o[(size_t)row0 * 4096 + n]               = acc00[r] * scale;
        o[(size_t)row0 * 4096 + 16 + n]          = acc01[r] * scale;
        o[(size_t)(row0 + 16) * 4096 + n]        = acc10[r] * scale;
        o[(size_t)(row0 + 16) * 4096 + 16 + n]   = acc11[r] * scale;
    }
}

// One block per row: register-cached values, LDS tree reductions, causal zeros.
__global__ __launch_bounds__(256) void softmax_causal(float* __restrict__ out)
{
    const int row = blockIdx.x;                // b*4096 + i
    const int i   = row & 4095;
    float* p = out + (size_t)row * 4096;
    const int tid = threadIdx.x;

    float vals[16];
    float mx = -INFINITY;
#pragma unroll
    for (int idx = 0; idx < 16; ++idx) {
        int   j = tid + (idx << 8);
        float v = (j <= i) ? p[j] : -INFINITY;
        vals[idx] = v;
        mx = fmaxf(mx, v);
    }

    __shared__ float red[256];
    red[tid] = mx;
    __syncthreads();
    for (int off = 128; off > 0; off >>= 1) {
        if (tid < off) red[tid] = fmaxf(red[tid], red[tid + off]);
        __syncthreads();
    }
    mx = red[0];
    __syncthreads();

    float sum = 0.f;
#pragma unroll
    for (int idx = 0; idx < 16; ++idx) {
        float e = __expf(vals[idx] - mx);      // exp(-inf)=0 above diagonal
        vals[idx] = e;
        sum += e;
    }
    red[tid] = sum;
    __syncthreads();
    for (int off = 128; off > 0; off >>= 1) {
        if (tid < off) red[tid] += red[tid + off];
        __syncthreads();
    }
    float inv = 1.0f / red[0];
#pragma unroll
    for (int idx = 0; idx < 16; ++idx) {
        int j = tid + (idx << 8);
        p[j] = vals[idx] * inv;                // j>i -> writes 0
    }
}

extern "C" void kernel_launch(void* const* d_in, const int* in_sizes, int n_in,
                              void* d_out, int out_size, void* d_ws, size_t ws_size,
                              hipStream_t stream) {
    const float* hid  = (const float*)d_in[0];  // [4,4096,512]
    const float* Wq   = (const float*)d_in[1];  // [512,512]
    const float* Wk   = (const float*)d_in[2];  // [512,512]
    const float* cosb = (const float*)d_in[3];  // [4096,512]
    const float* sinb = (const float*)d_in[4];  // [4096,512]
    float* out = (float*)d_out;                 // [4,4096,4096]

    const size_t NH = (size_t)16384 * 512;      // 8,388,608
    const size_t NW = (size_t)512 * 512;        // 262,144

    _Float16* h_h  = (_Float16*)d_ws;
    _Float16* wq_h = h_h + NH;
    _Float16* wk_h = wq_h + NW;
    _Float16* qh   = wk_h + NW;
    _Float16* kh   = qh + NH;                   // total ~51.4 MB of ws

    cvt_f32_f16<<<(unsigned)((NH + 255) / 256), 256, 0, stream>>>(hid, h_h, (int)NH);
    cvt_f32_f16<<<(unsigned)((NW + 255) / 256), 256, 0, stream>>>(Wq, wq_h, (int)NW);
    cvt_f32_f16<<<(unsigned)((NW + 255) / 256), 256, 0, stream>>>(Wk, wk_h, (int)NW);

    qk_proj_rope<<<dim3(16, 1024), 128, 0, stream>>>(h_h, wq_h, wk_h, cosb, sinb, qh, kh);

    scores_kernel<<<dim3(32, 64, 4), 256, 0, stream>>>(qh, kh, out);

    softmax_causal<<<16384, 256, 0, stream>>>(out);
}
